// TransformerLayerA_19104014532985
// MI455X (gfx1250) — matrix operand order repaired and checked by reading, not compiled, not run on hardware
//
#include <hip/hip_runtime.h>
#include <hip/hip_bf16.h>

// ---------------------------------------------------------------------------
// Types for CDNA5 WMMA (wave32): A/B = 16 bf16 per lane, C/D = 8 f32 per lane
// ---------------------------------------------------------------------------
typedef __attribute__((ext_vector_type(16))) __bf16 v16bf;
typedef __attribute__((ext_vector_type(8)))  __bf16 v8bf;
typedef __attribute__((ext_vector_type(8)))  float  v8f;

union BF16x16 { v16bf v; v8bf h[2]; };

// Load a 16-element bf16 fragment from two 8-element (16-byte) chunks.
__device__ __forceinline__ v16bf load_frag(const __bf16* p0, const __bf16* p1) {
    BF16x16 u;
    u.h[0] = *(const v8bf*)p0;
    u.h[1] = *(const v8bf*)p1;
    return u.v;
}

// Fixed problem dims
#define NB 4
#define TT 2048
#define DD 1024
#define HH 16
#define HD 64
#define FFD 4096
#define MM (NB * TT)   // 8192 rows

// ---------------------------------------------------------------------------
// fp32 -> bf16 convert
// ---------------------------------------------------------------------------
__global__ void tl_cvt_bf16(const float* __restrict__ src, __bf16* __restrict__ dst, int n) {
    int i = blockIdx.x * blockDim.x + threadIdx.x;
    if (i < n) dst[i] = (__bf16)src[i];
}

// concat(kq, v) -> X bf16  [8192, 1024]
__global__ void tl_concat_bf16(const float* __restrict__ kq, const float* __restrict__ v,
                               __bf16* __restrict__ X) {
    int i = blockIdx.x * blockDim.x + threadIdx.x;   // over 8192*1024
    int m = i >> 10;
    int c = i & 1023;
    float x = (c < 512) ? kq[(size_t)m * 512 + c] : v[(size_t)m * 512 + (c - 512)];
    X[i] = (__bf16)x;
}

// Branch-free tanh-form GELU (no exec-mask divergence in the epilogue).
__device__ __forceinline__ float gelu_fast(float x) {
    const float u  = 0.7978845608028654f * (x + 0.044715f * x * x * x);
    const float t  = __expf(-2.0f * fabsf(u));
    float th = (1.0f - t) / (1.0f + t);
    th = copysignf(th, u);
    return 0.5f * x * (1.0f + th);
}

// ---------------------------------------------------------------------------
// Tiled WMMA GEMM:  C[M,N] = A[M,K] * B[N,K]^T + bias
//   block tile 128x128x32, 256 threads = 8 waves, wave tile 32x64 (2x4 WMMA)
//   Double-buffered LDS (ping-pong): regs prefetch of tile t+1 overlaps the
//   WMMA work on tile t; single barrier per K-step.
// Epilogue modes:
//   0: obf  = bf16(C)                                   (Q, K projections)
//   1: of32 = C (Vflat) and obf = Vt bf16 [n*H, 64, T]  (V projection)
//   2: obf  = bf16(gelu(C))                             (FF1)
//   3: of32 = C + resid                                 (FF2; bias b2 + residual h)
// ---------------------------------------------------------------------------
#define BM 128
#define BN 128
#define BK 32
#define LDSP 40   // padded LDS row stride (elements); 80B rows keep b128 16B-aligned

__global__ __launch_bounds__(256)
void tl_gemm_wmma(const __bf16* __restrict__ A, const __bf16* __restrict__ B,
                  const float* __restrict__ bias, int M, int N, int K, int mode,
                  __bf16* __restrict__ obf, float* __restrict__ of32,
                  const float* __restrict__ resid) {
    __shared__ __bf16 sA[2][BM * LDSP];
    __shared__ __bf16 sB[2][BN * LDSP];

    const int tid  = threadIdx.x;
    const int lane = tid & 31;
    const int w    = tid >> 5;       // 0..7
    const int wr   = w >> 1;         // 0..3  -> m within block (x32)
    const int wc   = w & 1;          // 0..1  -> n within block (x64)
    const int m0   = blockIdx.y * BM;
    const int n0   = blockIdx.x * BN;

    const int lrow = tid >> 1;            // 0..127 cooperative-load row
    const int lcol = (tid & 1) * 16;      // 0 / 16

    const int fr   = lane & 15;           // fragment row (A) / col (B)
    const int kh8  = (lane >> 4) * 8;     // A/B-frag k offset per lane half

    const __bf16* ga = A + (size_t)(m0 + lrow) * K + lcol;
    const __bf16* gb = B + (size_t)(n0 + lrow) * K + lcol;
    const int lofs = lrow * LDSP + lcol;

    v8f acc[2][4] = {};

    // Prologue: stage tile 0 into LDS buffer 0.
    v8bf ra0 = *(const v8bf*)(ga);
    v8bf ra1 = *(const v8bf*)(ga + 8);
    v8bf rb0 = *(const v8bf*)(gb);
    v8bf rb1 = *(const v8bf*)(gb + 8);
    *(v8bf*)&sA[0][lofs]     = ra0;
    *(v8bf*)&sA[0][lofs + 8] = ra1;
    *(v8bf*)&sB[0][lofs]     = rb0;
    *(v8bf*)&sB[0][lofs + 8] = rb1;
    __syncthreads();

    int buf = 0;
    for (int k0 = 0; k0 < K; k0 += BK) {
        const bool more = (k0 + BK) < K;
        if (more) {
            // Prefetch next tile into registers (overlaps with WMMA below).
            ra0 = *(const v8bf*)(ga + k0 + BK);
            ra1 = *(const v8bf*)(ga + k0 + BK + 8);
            rb0 = *(const v8bf*)(gb + k0 + BK);
            rb1 = *(const v8bf*)(gb + k0 + BK + 8);
            if (k0 + 2 * BK < K) {   // warm L2 one more tile ahead
                __builtin_prefetch(ga + k0 + 2 * BK, 0, 1);
                __builtin_prefetch(gb + k0 + 2 * BK, 0, 1);
            }
        }

        v16bf af[2], bfr[4];
#pragma unroll
        for (int mi = 0; mi < 2; ++mi) {
            const __bf16* p = &sA[buf][(wr * 32 + mi * 16 + fr) * LDSP + kh8];
            af[mi] = load_frag(p, p + 16);
        }
#pragma unroll
        for (int ni = 0; ni < 4; ++ni) {
            const __bf16* p = &sB[buf][(wc * 64 + ni * 16 + fr) * LDSP + kh8];
            bfr[ni] = load_frag(p, p + 16);
        }
#pragma unroll
        for (int mi = 0; mi < 2; ++mi)
#pragma unroll
            for (int ni = 0; ni < 4; ++ni)
                acc[mi][ni] = __builtin_amdgcn_wmma_f32_16x16x32_bf16(
                    false, af[mi], false, bfr[ni], (short)0, acc[mi][ni], false, false);

        if (more) {
            const int nb = buf ^ 1;
            *(v8bf*)&sA[nb][lofs]     = ra0;
            *(v8bf*)&sA[nb][lofs + 8] = ra1;
            *(v8bf*)&sB[nb][lofs]     = rb0;
            *(v8bf*)&sB[nb][lofs + 8] = rb1;
            __syncthreads();
            buf = nb;
        }
    }

    const int rbase = (lane < 16) ? 0 : 8;
#pragma unroll
    for (int mi = 0; mi < 2; ++mi) {
#pragma unroll
        for (int ni = 0; ni < 4; ++ni) {
            const int col = n0 + wc * 64 + ni * 16 + fr;
            const float bv = bias[col];
#pragma unroll
            for (int i = 0; i < 8; ++i) {
                const int row = m0 + wr * 32 + mi * 16 + i + rbase;
                float val = acc[mi][ni][i] + bv;
                const size_t idx = (size_t)row * N + col;
                if (mode == 0) {
                    obf[idx] = (__bf16)val;
                } else if (mode == 1) {
                    of32[idx] = val;
                    const int b = row >> 11, t = row & 2047;   // row = n*2048 + t
                    const int h = col >> 6,  d = col & 63;     // col = h*64 + d
                    obf[(((size_t)(b * HH + h)) * HD + d) * TT + t] = (__bf16)val;
                } else if (mode == 2) {
                    obf[idx] = (__bf16)gelu_fast(val);
                } else {
                    of32[idx] = val + resid[idx];
                }
            }
        }
    }
}

// ---------------------------------------------------------------------------
// Causal flash attention, no 1/sqrt(hd) scaling (matches reference).
// Q,K: bf16 row-major [n*T, D] (head offset h*64). Vt: bf16 [n*H, 64, T].
// Output: f32 attn_out [n*T, D].
// Block = 128 threads (4 waves); wave w owns query rows q0..q0+15.
// ---------------------------------------------------------------------------
__global__ __launch_bounds__(128)
void tl_attn_wmma(const __bf16* __restrict__ Q, const __bf16* __restrict__ Kmat,
                  const __bf16* __restrict__ Vt, float* __restrict__ attnOut) {
    __shared__ __bf16 sP[4][16 * LDSP];  // per-wave P scratch (16 x 32, padded)

    const int tid  = threadIdx.x;
    const int lane = tid & 31;
    const int w    = tid >> 5;
    const int nh   = blockIdx.y;           // n*H + h
    const int n    = nh >> 4, h = nh & 15;
    const int q0   = blockIdx.x * 64 + w * 16;
    const int fr   = lane & 15;
    const int kh8  = (lane >> 4) * 8;

    const int rbase = (lane < 16) ? 0 : 8;

    const __bf16* Qb  = Q    + ((size_t)n * TT) * DD + h * HD;
    const __bf16* Kb  = Kmat + ((size_t)n * TT) * DD + h * HD;
    const __bf16* Vtb = Vt   + (size_t)nh * HD * TT;

    // Q fragments: rows q0..q0+15, feature halves [0,32) and [32,64)
    v16bf qf[2];
    {
        const __bf16* qrow = Qb + (size_t)(q0 + fr) * DD;
#pragma unroll
        for (int fh = 0; fh < 2; ++fh) {
            const __bf16* p = qrow + fh * 32 + kh8;
            qf[fh] = load_frag(p, p + 16);
        }
    }

    v8f o[4] = {};              // 16 x 64 output accumulator (f32)
    float m_i[8], l_i[8];
#pragma unroll
    for (int i = 0; i < 8; ++i) { m_i[i] = -INFINITY; l_i[i] = 0.0f; }

    const int qHi = q0 + 15;
    for (int k0 = 0; k0 <= qHi; k0 += 32) {
        // S = Q * K^T for 32 keys -> two 16x16 tiles
        v8f s[2] = {};
#pragma unroll
        for (int kt = 0; kt < 2; ++kt) {
            const int j = k0 + kt * 16 + fr;
            const __bf16* krow = Kb + (size_t)j * DD;
#pragma unroll
            for (int fh = 0; fh < 2; ++fh) {
                const __bf16* p = krow + fh * 32 + kh8;
                v16bf kf = load_frag(p, p + 16);
                s[kt] = __builtin_amdgcn_wmma_f32_16x16x32_bf16(
                    false, qf[fh], false, kf, (short)0, s[kt], false, false);
            }
        }

        // causal mask + online softmax (per-row stats live in 16-lane halves)
#pragma unroll
        for (int i = 0; i < 8; ++i) {
            const int r  = q0 + i + rbase;
            const int j0 = k0 + fr, j1 = k0 + 16 + fr;
            float s0 = (j0 <= r) ? s[0][i] : -INFINITY;
            float s1 = (j1 <= r) ? s[1][i] : -INFINITY;
            float mx = fmaxf(s0, s1);
#pragma unroll
            for (int d = 8; d >= 1; d >>= 1) mx = fmaxf(mx, __shfl_xor(mx, d, 32));
            const float mnew  = fmaxf(m_i[i], mx);
            const float scale = __expf(m_i[i] - mnew);
            const float p0 = __expf(s0 - mnew);
            const float p1 = __expf(s1 - mnew);
            float ps = p0 + p1;
#pragma unroll
            for (int d = 8; d >= 1; d >>= 1) ps += __shfl_xor(ps, d, 32);
            l_i[i] = l_i[i] * scale + ps;
            m_i[i] = mnew;
#pragma unroll
            for (int fi = 0; fi < 4; ++fi) o[fi][i] *= scale;
            // stash P (bf16) to LDS in A-matrix memory order [row][k]
            sP[w][(i + rbase) * LDSP + fr]      = (__bf16)p0;
            sP[w][(i + rbase) * LDSP + 16 + fr] = (__bf16)p1;
        }

        // P as A-fragment (LDS ops are in-order within a wave)
        const __bf16* pp = &sP[w][fr * LDSP + kh8];
        v16bf pf = load_frag(pp, pp + 16);

        // O += P (16x32) * V (32x64) via transposed V
#pragma unroll
        for (int fi = 0; fi < 4; ++fi) {
            const __bf16* vp = Vtb + (size_t)(fi * 16 + fr) * TT + k0 + kh8;
            v16bf vf = load_frag(vp, vp + 16);
            o[fi] = __builtin_amdgcn_wmma_f32_16x16x32_bf16(
                false, pf, false, vf, (short)0, o[fi], false, false);
        }
    }

    float rl[8];
#pragma unroll
    for (int i = 0; i < 8; ++i) rl[i] = 1.0f / l_i[i];
#pragma unroll
    for (int fi = 0; fi < 4; ++fi)
#pragma unroll
        for (int i = 0; i < 8; ++i) {
            const int r = q0 + i + rbase;
            attnOut[((size_t)(n * TT + r)) * DD + h * HD + fi * 16 + fr] = o[fi][i] * rl[i];
        }
}

// ---------------------------------------------------------------------------
// x = a (+ b);  y = LayerNorm(x) * g + beta.  One block (256 thr) per row.
// Writes f32 and optionally bf16.
// ---------------------------------------------------------------------------
__global__ __launch_bounds__(256)
void tl_add_ln(const float* __restrict__ a, const float* __restrict__ b,
               const float* __restrict__ g, const float* __restrict__ be,
               float* __restrict__ outF, __bf16* __restrict__ outBf) {
    const int row = blockIdx.x;
    const int tid = threadIdx.x;
    const float* pa = a + (size_t)row * DD;
    const float* pb = b ? (b + (size_t)row * DD) : nullptr;

    float x[4];
    float s = 0.0f, sq = 0.0f;
#pragma unroll
    for (int i = 0; i < 4; ++i) {
        const int c = tid + i * 256;
        x[i] = pb ? (pa[c] + pb[c]) : pa[c];
        s += x[i]; sq += x[i] * x[i];
    }
#pragma unroll
    for (int d = 16; d >= 1; d >>= 1) { s += __shfl_xor(s, d, 32); sq += __shfl_xor(sq, d, 32); }

    __shared__ float ws[8], wq[8];
    if ((tid & 31) == 0) { ws[tid >> 5] = s; wq[tid >> 5] = sq; }
    __syncthreads();
    s = 0.0f; sq = 0.0f;
#pragma unroll
    for (int i = 0; i < 8; ++i) { s += ws[i]; sq += wq[i]; }

    const float mean = s * (1.0f / DD);
    const float var  = sq * (1.0f / DD) - mean * mean;
    const float rstd = rsqrtf(var + 1e-5f);
#pragma unroll
    for (int i = 0; i < 4; ++i) {
        const int c = tid + i * 256;
        const float y = (x[i] - mean) * rstd * g[c] + be[c];
        outF[(size_t)row * DD + c] = y;
        if (outBf) outBf[(size_t)row * DD + c] = (__bf16)y;
    }
}

// ---------------------------------------------------------------------------
// Launcher
// ---------------------------------------------------------------------------
extern "C" void kernel_launch(void* const* d_in, const int* in_sizes, int n_in,
                              void* d_out, int out_size, void* d_ws, size_t ws_size,
                              hipStream_t stream) {
    const float* kq  = (const float*)d_in[0];
    const float* v   = (const float*)d_in[1];
    const float* Wk  = (const float*)d_in[2];
    const float* bk  = (const float*)d_in[3];
    const float* Wq  = (const float*)d_in[4];
    const float* bq  = (const float*)d_in[5];
    const float* Wv  = (const float*)d_in[6];
    const float* bv  = (const float*)d_in[7];
    const float* W1  = (const float*)d_in[8];
    const float* b1  = (const float*)d_in[9];
    const float* W2  = (const float*)d_in[10];
    const float* b2  = (const float*)d_in[11];
    const float* g1  = (const float*)d_in[12];
    const float* be1 = (const float*)d_in[13];
    const float* g2  = (const float*)d_in[14];
    const float* be2 = (const float*)d_in[15];
    float* out = (float*)d_out;

    char* ws = (char*)d_ws;
    const size_t MB = 1024ull * 1024ull;
    __bf16* wWk = (__bf16*)(ws + 0 * MB);     // 2 MB
    __bf16* wWq = (__bf16*)(ws + 2 * MB);     // 2 MB
    __bf16* wWv = (__bf16*)(ws + 4 * MB);     // 2 MB
    __bf16* wW1 = (__bf16*)(ws + 6 * MB);     // 8 MB
    __bf16* wW2 = (__bf16*)(ws + 14 * MB);    // 8 MB
    __bf16* Xbf = (__bf16*)(ws + 22 * MB);    // 16 MB
    __bf16* Qbf = (__bf16*)(ws + 38 * MB);    // 16 MB
    __bf16* Kbf = (__bf16*)(ws + 54 * MB);    // 16 MB
    __bf16* Vtb = (__bf16*)(ws + 70 * MB);    // 16 MB  [n*H, 64, T]
    float*  Vfl = (float*)(ws + 86 * MB);     // 32 MB
    float*  atn = (float*)(ws + 118 * MB);    // 32 MB
    float*  hF  = (float*)(ws + 150 * MB);    // 32 MB
    __bf16* hBf = (__bf16*)(ws + 182 * MB);   // 16 MB
    __bf16* ff1 = (__bf16*)(ws + 22 * MB);    // 64 MB, reuses X/Q/K/Vt (dead after attention)
    float*  yF  = (float*)(ws + 86 * MB);     // 32 MB, reuses Vflat (dead after LN1)

    // 1) precision conversion
    tl_cvt_bf16<<<(DD * DD) / 256, 256, 0, stream>>>(Wk, wWk, DD * DD);
    tl_cvt_bf16<<<(DD * DD) / 256, 256, 0, stream>>>(Wq, wWq, DD * DD);
    tl_cvt_bf16<<<(DD * DD) / 256, 256, 0, stream>>>(Wv, wWv, DD * DD);
    tl_cvt_bf16<<<(FFD * DD) / 256, 256, 0, stream>>>(W1, wW1, FFD * DD);
    tl_cvt_bf16<<<(DD * FFD) / 256, 256, 0, stream>>>(W2, wW2, DD * FFD);
    tl_concat_bf16<<<(MM * DD) / 256, 256, 0, stream>>>(kq, v, Xbf);

    // 2) QKV projections
    dim3 gQKV(DD / BN, MM / BM);
    tl_gemm_wmma<<<gQKV, 256, 0, stream>>>(Xbf, wWq, bq, MM, DD, DD, 0, Qbf, nullptr, nullptr);
    tl_gemm_wmma<<<gQKV, 256, 0, stream>>>(Xbf, wWk, bk, MM, DD, DD, 0, Kbf, nullptr, nullptr);
    tl_gemm_wmma<<<gQKV, 256, 0, stream>>>(Xbf, wWv, bv, MM, DD, DD, 1, Vtb, Vfl, nullptr);

    // 3) causal attention
    tl_attn_wmma<<<dim3(TT / 64, NB * HH), 128, 0, stream>>>(Qbf, Kbf, Vtb, atn);

    // 4) h = LN(Vflat + attn)
    tl_add_ln<<<MM, 256, 0, stream>>>(Vfl, atn, g1, be1, hF, hBf);

    // 5) FFN
    tl_gemm_wmma<<<dim3(FFD / BN, MM / BM), 256, 0, stream>>>(hBf, wW1, b1, MM, FFD, DD, 2,
                                                              ff1, nullptr, nullptr);
    tl_gemm_wmma<<<dim3(DD / BN, MM / BM), 256, 0, stream>>>(ff1, wW2, b2, MM, DD, FFD, 3,
                                                             nullptr, yF, hF);

    // 6) out = LN(h + ff)   (yF already holds the sum)
    tl_add_ln<<<MM, 256, 0, stream>>>(yF, nullptr, g2, be2, out, nullptr);
}
